// SNet_fgn_ver1_39032662786625
// MI455X (gfx1250) — matrix operand order repaired and checked by reading, not compiled, not run on hardware
//
#include <hip/hip_runtime.h>

typedef _Float16 half_t;
typedef __attribute__((ext_vector_type(16))) _Float16 v16h;
typedef __attribute__((ext_vector_type(8)))  float    v8f;

#define EPS 1e-5f

// ---------------------------------------------------------------------------
// Fragment loaders (CDNA5 wave32 WMMA 16x16x32 f16 layouts, CDNA5 ISA)
// A (16x32, MxK): lane L: M=L&15; element i: K = (i&8)*2 + (L>>4)*8 + (i&7)
//   -> per lane two contiguous 8-half chunks at c0+(half*8) and c0+16+(half*8)
// B (32x16, KxN): lane L: N=L&15; element i: K = (i&8)*2 + (L>>4)*8 + (i&7)  (pre-packed)
// C/D (16x16 f32): lane L: N=L&15; vgpr v: M = v + 8*(L>>4)
// ---------------------------------------------------------------------------
__device__ __forceinline__ v16h load_A_frag(const half_t* __restrict__ in, int b, int y,
                                            int x0, int H, int W, int Cin, int tap, int cbase) {
    int lane = threadIdx.x & 31;
    int halfid = lane >> 4;
    int m = lane & 15;
    int yy = y + tap / 3 - 1;
    int xx = x0 + m + tap % 3 - 1;
    union { v16h h; uint4 u[2]; } U;
    U.u[0] = make_uint4(0u, 0u, 0u, 0u);
    U.u[1] = make_uint4(0u, 0u, 0u, 0u);
    if (yy >= 0 && yy < H && xx >= 0 && xx < W) {
        const half_t* p = in + (((size_t)b * H + yy) * W + xx) * Cin + cbase + (halfid << 3);
        U.u[0] = *(const uint4*)(p);        // c = cbase + half*8 + 0..7
        U.u[1] = *(const uint4*)(p + 16);   // c = cbase + 16 + half*8 + 0..7
    }
    return U.h;
}

// unguarded loader: per-lane base pointer + uniform (compile-time) offset
__device__ __forceinline__ v16h load_A_fast(const half_t* __restrict__ p, int off) {
    union { v16h h; uint4 u[2]; } U;
    U.u[0] = *(const uint4*)(p + off);
    U.u[1] = *(const uint4*)(p + off + 16);
    return U.h;
}

__device__ __forceinline__ v16h load_B_lds(const half_t* Bl, int ks, int nt, int NT) {
    int lane = threadIdx.x & 31;
    const half_t* p = Bl + (((ks * NT + nt) * 32 + lane) << 4);
    union { v16h h; uint4 u[2]; } U;
    U.u[0] = *(const uint4*)(p);
    U.u[1] = *(const uint4*)(p + 8);
    return U.h;
}

// ---------------------------------------------------------------------------
// Weight pack: raw OIHW f32 -> fragment-ordered f16 for tpn1 (K=576,N=64) and
// df1 (K=288, N padded 81->96). Also pads df1 bias to 96.
// ---------------------------------------------------------------------------
__global__ void pack_weights_kernel(const float* __restrict__ w1,   // [64][64][3][3]
                                    const float* __restrict__ w2,   // [81][32][3][3]
                                    const float* __restrict__ b2,   // [81]
                                    half_t* __restrict__ B1, half_t* __restrict__ B2,
                                    float* __restrict__ bias96) {
    int idx = blockIdx.x * blockDim.x + threadIdx.x;
    if (idx < 18 * 4 * 32 * 16) {
        int i = idx & 15, lane = (idx >> 4) & 31, rest = idx >> 9;
        int nt = rest & 3, ks = rest >> 2;
        int k = ks * 32 + (i & 8) * 2 + (lane >> 4) * 8 + (i & 7);      // k = tap*64 + c
        int n = nt * 16 + (lane & 15);
        int tap = k / 64, c = k % 64;
        B1[idx] = (half_t)w1[(n * 64 + c) * 9 + tap];
    }
    if (idx < 9 * 6 * 32 * 16) {
        int i = idx & 15, lane = (idx >> 4) & 31, rest = idx >> 9;
        int nt = rest % 6, ks = rest / 6;
        int k = ks * 32 + (i & 8) * 2 + (lane >> 4) * 8 + (i & 7);      // k = tap*32 + c
        int n = nt * 16 + (lane & 15);
        int tap = k / 32, c = k % 32;
        B2[idx] = (n < 81) ? (half_t)w2[(n * 32 + c) * 9 + tap] : (half_t)0.f;
    }
    if (idx < 96) bias96[idx] = (idx < 81) ? b2[idx] : 0.f;
}

// ---------------------------------------------------------------------------
// Small f32 planar conv3x3 (+BN fold +act) for the conv_g chain (<3% of FLOPs)
// ACT: 0 none, 1 relu, 2 sigmoid
// ---------------------------------------------------------------------------
template <int CIN, int COUT, int ACT>
__global__ void conv3x3_f32(const float* __restrict__ in, float* __restrict__ out,
                            const float* __restrict__ w, const float* __restrict__ bias,
                            const float* __restrict__ gm, const float* __restrict__ bt,
                            const float* __restrict__ mu, const float* __restrict__ vr,
                            int BB, int H, int W) {
    __shared__ float wf[COUT * CIN * 9];
    __shared__ float bf[COUT];
    for (int t = threadIdx.x; t < COUT * CIN * 9; t += blockDim.x) {
        int co = t / (CIN * 9);
        float s = gm[co] * rsqrtf(vr[co] + EPS);
        wf[t] = w[t] * s;
    }
    for (int t = threadIdx.x; t < COUT; t += blockDim.x) {
        float s = gm[t] * rsqrtf(vr[t] + EPS);
        bf[t] = (bias[t] - mu[t]) * s + bt[t];
    }
    __syncthreads();
    int idx = blockIdx.x * blockDim.x + threadIdx.x;
    int total = BB * H * W;
    if (idx >= total) return;
    int x = idx % W, y = (idx / W) % H, b = idx / (W * H);
    float acc[COUT];
#pragma unroll
    for (int c = 0; c < COUT; c++) acc[c] = bf[c];
    for (int ci = 0; ci < CIN; ci++) {
        const float* ip = in + ((size_t)b * CIN + ci) * H * W;
#pragma unroll
        for (int kh = 0; kh < 3; kh++) {
            int yy = y + kh - 1;
            if (yy < 0 || yy >= H) continue;
#pragma unroll
            for (int kw = 0; kw < 3; kw++) {
                int xx = x + kw - 1;
                if (xx < 0 || xx >= W) continue;
                float v = ip[yy * W + xx];
#pragma unroll
                for (int co = 0; co < COUT; co++)
                    acc[co] += v * wf[(co * CIN + ci) * 9 + kh * 3 + kw];
            }
        }
    }
#pragma unroll
    for (int co = 0; co < COUT; co++) {
        float v = acc[co];
        if (ACT == 1) v = fmaxf(v, 0.f);
        if (ACT == 2) v = 1.f / (1.f + expf(-v));
        out[((size_t)b * COUT + co) * H * W + y * W + x] = v;
    }
}

// ---------------------------------------------------------------------------
// conv3x3 1->C, ReLU, f32 planar in -> f16 NHWC out  (tpn0: C=64, df0: C=32)
// ---------------------------------------------------------------------------
template <int C>
__global__ void conv1toC_nhwc(const float* __restrict__ in, half_t* __restrict__ out,
                              const float* __restrict__ w, const float* __restrict__ bias,
                              int BB, int H, int W) {
    __shared__ float wl[C * 9];
    __shared__ float bl[C];
    for (int t = threadIdx.x; t < C * 9; t += blockDim.x) wl[t] = w[t];
    for (int t = threadIdx.x; t < C; t += blockDim.x) bl[t] = bias[t];
    __syncthreads();
    int idx = blockIdx.x * blockDim.x + threadIdx.x;
    int total = BB * H * W;
    if (idx >= total) return;
    int x = idx % W, y = (idx / W) % H, b = idx / (W * H);
    const float* ip = in + (size_t)b * H * W;
    float px[9];
#pragma unroll
    for (int t = 0; t < 9; t++) {
        int yy = y + t / 3 - 1, xx = x + t % 3 - 1;
        px[t] = (yy >= 0 && yy < H && xx >= 0 && xx < W) ? ip[yy * W + xx] : 0.f;
    }
    uint* op = (uint*)(out + (size_t)idx * C);
#pragma unroll
    for (int c = 0; c < C; c += 2) {
        float a0 = bl[c], a1 = bl[c + 1];
#pragma unroll
        for (int t = 0; t < 9; t++) {
            a0 += px[t] * wl[c * 9 + t];
            a1 += px[t] * wl[(c + 1) * 9 + t];
        }
        union { uint u; half_t h[2]; } P;
        P.h[0] = (half_t)fmaxf(a0, 0.f);
        P.h[1] = (half_t)fmaxf(a1, 0.f);
        op[c >> 1] = P.u;
    }
}

// ---------------------------------------------------------------------------
// tpn1: 64->64 conv3x3 + bias + ReLU via WMMA.
// 1 wave = 32 pixels (2 M-tiles); weights staged in LDS, shared by 4 waves.
// Interior waves (~79%): fully unrolled loop, unguarded constant-offset loads.
// Boundary waves: guarded, software-pipelined loop.
// K = 576 = 18 ksteps of 32;  N = 64 = 4 tiles of 16.
// ---------------------------------------------------------------------------
__global__ void __launch_bounds__(128) tpn1_wmma_kernel(
    const half_t* __restrict__ x1, const half_t* __restrict__ Bf,
    const float* __restrict__ bias, half_t* __restrict__ x2) {
    constexpr int H = 320, W = 320;
    __shared__ half_t Bl[18 * 4 * 32 * 16];      // 73,728 B
    {
        const uint4* s = (const uint4*)Bf;
        uint4* d = (uint4*)Bl;
        for (int i = threadIdx.x; i < (18 * 4 * 32 * 16) / 8; i += 128) d[i] = s[i];
    }
    __syncthreads();

    int wslot = threadIdx.x >> 5;
    int lane = threadIdx.x & 31;
    int halfid = lane >> 4;
    int n0 = lane & 15;
    int tilesPerRow = W >> 5;                    // 10 (32-pixel wave tiles)
    int wave = blockIdx.x * 4 + wslot;
    int xt = wave % tilesPerRow;
    int t = wave / tilesPerRow;
    int y = t % H, b = t / H;
    int x0 = xt << 5;

    v8f vzero = {0.f, 0.f, 0.f, 0.f, 0.f, 0.f, 0.f, 0.f};
    v8f acc[2][4];
#pragma unroll
    for (int mt = 0; mt < 2; mt++)
#pragma unroll
        for (int nt = 0; nt < 4; nt++) acc[mt][nt] = vzero;

    bool interior = (y >= 1) && (y <= H - 2) && (x0 >= 1) && (x0 <= W - 33);
    if (interior) {
        const half_t* aB0 = x1 + (((size_t)b * H + y) * W + x0 + (lane & 15)) * 64 + (halfid << 3);
        const half_t* aB1 = aB0 + (size_t)16 * 64;
#pragma unroll
        for (int ks = 0; ks < 18; ks++) {
            int tap = ks >> 1;
            int off = ((tap / 3 - 1) * W + (tap % 3 - 1)) * 64 + ((ks & 1) << 5); // compile-time
            v16h a0 = load_A_fast(aB0, off);
            v16h a1 = load_A_fast(aB1, off);
            v16h bb[4];
#pragma unroll
            for (int nt = 0; nt < 4; nt++) bb[nt] = load_B_lds(Bl, ks, nt, 4);
#pragma unroll
            for (int nt = 0; nt < 4; nt++) {
                acc[0][nt] = __builtin_amdgcn_wmma_f32_16x16x32_f16(
                    false, a0, false, bb[nt], (short)0, acc[0][nt], false, false);
                acc[1][nt] = __builtin_amdgcn_wmma_f32_16x16x32_f16(
                    false, a1, false, bb[nt], (short)0, acc[1][nt], false, false);
            }
        }
    } else {
        v16h aC[2], aN[2];
        aC[0] = load_A_frag(x1, b, y, x0, H, W, 64, 0, 0);
        aC[1] = load_A_frag(x1, b, y, x0 + 16, H, W, 64, 0, 0);
        for (int ks = 0; ks < 18; ks++) {
            if (ks < 17) {
                int tap = (ks + 1) >> 1;
                int cb = ((ks + 1) & 1) << 5;
                aN[0] = load_A_frag(x1, b, y, x0, H, W, 64, tap, cb);
                aN[1] = load_A_frag(x1, b, y, x0 + 16, H, W, 64, tap, cb);
            }
            v16h bb[4];
#pragma unroll
            for (int nt = 0; nt < 4; nt++) bb[nt] = load_B_lds(Bl, ks, nt, 4);
#pragma unroll
            for (int nt = 0; nt < 4; nt++)
#pragma unroll
                for (int mt = 0; mt < 2; mt++)
                    acc[mt][nt] = __builtin_amdgcn_wmma_f32_16x16x32_f16(
                        false, aC[mt], false, bb[nt], (short)0, acc[mt][nt], false, false);
            aC[0] = aN[0];
            aC[1] = aN[1];
        }
    }
#pragma unroll
    for (int mt = 0; mt < 2; mt++)
#pragma unroll
        for (int nt = 0; nt < 4; nt++) {
            int n = nt * 16 + n0;
            float bv = bias[n];
#pragma unroll
            for (int v = 0; v < 8; v++) {
                int m = v + (halfid << 3);
                float val = fmaxf(acc[mt][nt][v] + bv, 0.f);
                x2[(((size_t)b * H + y) * W + (x0 + mt * 16 + m)) * 64 + n] = (half_t)val;
            }
        }
}

// ---------------------------------------------------------------------------
// tpn2: 64->1 conv3x3 (no act), f16 NHWC in -> f32 planar out
// ---------------------------------------------------------------------------
__global__ void tpn2_conv(const half_t* __restrict__ x2, float* __restrict__ x_,
                          const float* __restrict__ w, const float* __restrict__ bias,
                          int BB, int H, int W) {
    __shared__ float wl[64 * 9];
    for (int t = threadIdx.x; t < 64 * 9; t += blockDim.x) wl[t] = w[t];
    __syncthreads();
    int idx = blockIdx.x * blockDim.x + threadIdx.x;
    int total = BB * H * W;
    if (idx >= total) return;
    int x = idx % W, y = (idx / W) % H, b = idx / (W * H);
    float acc = bias[0];
    for (int tap = 0; tap < 9; tap++) {
        int yy = y + tap / 3 - 1, xx = x + tap % 3 - 1;
        if (yy < 0 || yy >= H || xx < 0 || xx >= W) continue;
        const uint4* q = (const uint4*)(x2 + (((size_t)b * H + yy) * W + xx) * 64);
#pragma unroll
        for (int g8 = 0; g8 < 8; g8++) {
            union { uint4 u; half_t h[8]; } U;
            U.u = q[g8];
#pragma unroll
            for (int j = 0; j < 8; j++)
                acc += (float)U.h[j] * wl[(g8 * 8 + j) * 9 + tap];
        }
    }
    x_[idx] = acc;
}

// ---------------------------------------------------------------------------
// c1: conv3x3 over concat(x_, g_) (2ch) -> 1ch, no act
// ---------------------------------------------------------------------------
__global__ void c1_conv(const float* __restrict__ x_, const float* __restrict__ g_,
                        float* __restrict__ xc, const float* __restrict__ w,
                        const float* __restrict__ bias, int BB, int H, int W) {
    int idx = blockIdx.x * blockDim.x + threadIdx.x;
    int total = BB * H * W;
    if (idx >= total) return;
    int x = idx % W, y = (idx / W) % H, b = idx / (W * H);
    const float* p0 = x_ + (size_t)b * H * W;
    const float* p1 = g_ + (size_t)b * H * W;
    float acc = bias[0];
#pragma unroll
    for (int tap = 0; tap < 9; tap++) {
        int yy = y + tap / 3 - 1, xx = x + tap % 3 - 1;
        if (yy < 0 || yy >= H || xx < 0 || xx >= W) continue;
        acc += p0[yy * W + xx] * w[tap] + p1[yy * W + xx] * w[9 + tap];
    }
    xc[idx] = acc;
}

// ---------------------------------------------------------------------------
// df1 (32->81 conv3x3) + softmax(81) + 9x9 kernel-prediction dot, fully fused.
// 1 wave = 32 pixels (2 M-tiles); weights in LDS; interior fast path.
// WMMA: K = 288 = 9 ksteps of 32; N padded to 96 = 6 tiles of 16.
// Phase 2: logits in LDS (row stride 97 => conflict-free column reads, padded
// channels 81..95 land harmlessly inside the row), each lane owns one pixel's
// softmax + patch dot. Logits never touch HBM.
// ---------------------------------------------------------------------------
__global__ void __launch_bounds__(128) df1_softmax_dot_kernel(
    const half_t* __restrict__ h32, const half_t* __restrict__ Bf,
    const float* __restrict__ bias96, const float* __restrict__ xc,
    float* __restrict__ out) {
    constexpr int H = 320, W = 320;
    __shared__ half_t Bl[9 * 6 * 32 * 16];       // 55,296 B
    __shared__ float lg[4][32][97];              // 49,664 B
    {
        const uint4* s = (const uint4*)Bf;
        uint4* d = (uint4*)Bl;
        for (int i = threadIdx.x; i < (9 * 6 * 32 * 16) / 8; i += 128) d[i] = s[i];
    }
    __syncthreads();

    int wslot = threadIdx.x >> 5;
    int lane = threadIdx.x & 31;
    int halfid = lane >> 4;
    int n0 = lane & 15;
    int tilesPerRow = W >> 5;
    int wave = blockIdx.x * 4 + wslot;
    int xt = wave % tilesPerRow;
    int t = wave / tilesPerRow;
    int y = t % H, b = t / H;
    int x0 = xt << 5;

    v8f vzero = {0.f, 0.f, 0.f, 0.f, 0.f, 0.f, 0.f, 0.f};
    v8f acc[2][6];
#pragma unroll
    for (int mt = 0; mt < 2; mt++)
#pragma unroll
        for (int nt = 0; nt < 6; nt++) acc[mt][nt] = vzero;

    bool interior = (y >= 1) && (y <= H - 2) && (x0 >= 1) && (x0 <= W - 33);
    if (interior) {
        const half_t* aB0 = h32 + (((size_t)b * H + y) * W + x0 + (lane & 15)) * 32 + (halfid << 3);
        const half_t* aB1 = aB0 + (size_t)16 * 32;
#pragma unroll
        for (int ks = 0; ks < 9; ks++) {
            int off = ((ks / 3 - 1) * W + (ks % 3 - 1)) * 32;   // compile-time
            v16h a0 = load_A_fast(aB0, off);
            v16h a1 = load_A_fast(aB1, off);
            v16h bb[6];
#pragma unroll
            for (int nt = 0; nt < 6; nt++) bb[nt] = load_B_lds(Bl, ks, nt, 6);
#pragma unroll
            for (int nt = 0; nt < 6; nt++) {
                acc[0][nt] = __builtin_amdgcn_wmma_f32_16x16x32_f16(
                    false, a0, false, bb[nt], (short)0, acc[0][nt], false, false);
                acc[1][nt] = __builtin_amdgcn_wmma_f32_16x16x32_f16(
                    false, a1, false, bb[nt], (short)0, acc[1][nt], false, false);
            }
        }
    } else {
        v16h aC[2], aN[2];
        aC[0] = load_A_frag(h32, b, y, x0, H, W, 32, 0, 0);
        aC[1] = load_A_frag(h32, b, y, x0 + 16, H, W, 32, 0, 0);
        for (int ks = 0; ks < 9; ks++) {
            if (ks < 8) {
                aN[0] = load_A_frag(h32, b, y, x0, H, W, 32, ks + 1, 0);
                aN[1] = load_A_frag(h32, b, y, x0 + 16, H, W, 32, ks + 1, 0);
            }
            v16h bb[6];
#pragma unroll
            for (int nt = 0; nt < 6; nt++) bb[nt] = load_B_lds(Bl, ks, nt, 6);
#pragma unroll
            for (int nt = 0; nt < 6; nt++)
#pragma unroll
                for (int mt = 0; mt < 2; mt++)
                    acc[mt][nt] = __builtin_amdgcn_wmma_f32_16x16x32_f16(
                        false, aC[mt], false, bb[nt], (short)0, acc[mt][nt], false, false);
            aC[0] = aN[0];
            aC[1] = aN[1];
        }
    }
    // scatter logits (+bias) to LDS; n in [81,95] is harmless in-row padding
#pragma unroll
    for (int mt = 0; mt < 2; mt++)
#pragma unroll
        for (int nt = 0; nt < 6; nt++) {
            int n = nt * 16 + n0;
            float bv = bias96[n];
#pragma unroll
            for (int v = 0; v < 8; v++) {
                int m = mt * 16 + v + (halfid << 3);
                lg[wslot][m][n] = acc[mt][nt][v] + bv;
            }
        }
    __syncthreads();

    // each lane owns pixel m = lane
    int px = x0 + lane;
    const float* lrow = lg[wslot][lane];
    float mx = -3.0e38f;
    for (int n = 0; n < 81; n++) mx = fmaxf(mx, lrow[n]);
    const float* xcb = xc + (size_t)b * H * W;
    float s = 0.f, dot = 0.f;
    if (y >= 4 && y <= H - 5 && x0 >= 4 && x0 <= W - 36) {     // full 9x9 in bounds
        const float* pb = xcb + (size_t)(y - 4) * W + (px - 4);
#pragma unroll
        for (int kh = 0; kh < 9; kh++)
#pragma unroll
            for (int kw = 0; kw < 9; kw++) {
                float e = expf(lrow[kh * 9 + kw] - mx);
                s += e;
                dot += e * pb[kh * W + kw];                     // constant offsets
            }
    } else {
#pragma unroll
        for (int kh = 0; kh < 9; kh++) {
            int yy = y + kh - 4;
            bool rowok = (yy >= 0 && yy < H);
#pragma unroll
            for (int kw = 0; kw < 9; kw++) {
                float e = expf(lrow[kh * 9 + kw] - mx);
                s += e;
                int xx = px + kw - 4;
                float v = (rowok && xx >= 0 && xx < W) ? xcb[yy * W + xx] : 0.f;
                dot += e * v;
            }
        }
    }
    out[((size_t)b * H + y) * W + px] = dot / s;
}

// ---------------------------------------------------------------------------
extern "C" void kernel_launch(void* const* d_in, const int* in_sizes, int n_in,
                              void* d_out, int out_size, void* d_ws, size_t ws_size,
                              hipStream_t stream) {
    const int B = 8, H = 320, W = 320;
    const float* x = (const float*)d_in[0];
    const float* g = (const float*)d_in[1];
    // params (flattened in setup_inputs insertion order)
    const float* cg0_w = (const float*)d_in[2];
    const float* cg0_b = (const float*)d_in[3];
    const float* cg0_gm = (const float*)d_in[4];
    const float* cg0_bt = (const float*)d_in[5];
    const float* cg0_mu = (const float*)d_in[6];
    const float* cg0_vr = (const float*)d_in[7];
    const float* cg1_w = (const float*)d_in[8];
    const float* cg1_b = (const float*)d_in[9];
    const float* cg1_gm = (const float*)d_in[10];
    const float* cg1_bt = (const float*)d_in[11];
    const float* cg1_mu = (const float*)d_in[12];
    const float* cg1_vr = (const float*)d_in[13];
    const float* cg2_w = (const float*)d_in[14];
    const float* cg2_b = (const float*)d_in[15];
    const float* cg2_gm = (const float*)d_in[16];
    const float* cg2_bt = (const float*)d_in[17];
    const float* cg2_mu = (const float*)d_in[18];
    const float* cg2_vr = (const float*)d_in[19];
    const float* cg3_w = (const float*)d_in[20];
    const float* cg3_b = (const float*)d_in[21];
    const float* cg3_gm = (const float*)d_in[22];
    const float* cg3_bt = (const float*)d_in[23];
    const float* cg3_mu = (const float*)d_in[24];
    const float* cg3_vr = (const float*)d_in[25];
    const float* tpn0_w = (const float*)d_in[26];
    const float* tpn0_b = (const float*)d_in[27];
    const float* tpn1_w = (const float*)d_in[28];
    const float* tpn1_b = (const float*)d_in[29];
    const float* tpn2_w = (const float*)d_in[30];
    const float* tpn2_b = (const float*)d_in[31];
    const float* c1_w = (const float*)d_in[32];
    const float* c1_b = (const float*)d_in[33];
    const float* df0_w = (const float*)d_in[34];
    const float* df0_b = (const float*)d_in[35];
    const float* df1_w = (const float*)d_in[36];
    const float* df1_b = (const float*)d_in[37];
    float* out = (float*)d_out;

    // workspace layout (regions aliased across stage lifetimes)
    const size_t PIX = (size_t)B * H * W;           // 819200
    const size_t R_BYTES = PIX * 64 * 2;            // 104,857,600
    char* ws = (char*)d_ws;
    half_t* B1 = (half_t*)(ws + 0);                 // 73,728 B
    half_t* B2 = (half_t*)(ws + 73728);             // 55,296 B
    float* bias96 = (float*)(ws + 129024);          // 384 B
    char* R1 = ws + 131072;                         // 105 MB: t1 -> x1 -> h32
    char* R2 = R1 + R_BYTES;                        // 105 MB: t0,t2 -> x2
    char* SM = R2 + R_BYTES;                        // smalls: g_, x_, xc
    float* t0 = (float*)R2;                         // 16ch f32 = 52.4 MB
    float* t1 = (float*)R1;                         //  8ch f32 = 26.2 MB
    float* t2 = (float*)(R2 + PIX * 16 * 4);        //  4ch f32 = 13.1 MB
    float* g_ = (float*)SM;
    float* x_ = (float*)(SM + PIX * 4);
    float* xc = (float*)(SM + 2 * PIX * 4);
    half_t* x1 = (half_t*)R1;
    half_t* x2 = (half_t*)R2;
    half_t* h32 = (half_t*)R1;

    dim3 blk(256);
    dim3 grd((unsigned)((PIX + 255) / 256));
    int nWaves = B * H * (W / 32);                  // 25600 (32-px wave tiles)
    dim3 wgrd((unsigned)(nWaves / 4));              // 4 waves/block -> 6400

    // 0) pack WMMA weight fragments
    pack_weights_kernel<<<144, 256, 0, stream>>>(tpn1_w, df1_w, df1_b, B1, B2, bias96);
    // 1) conv_g chain (f32 VALU, ~3% of FLOPs)
    conv3x3_f32<1, 16, 1><<<grd, blk, 0, stream>>>(g, t0, cg0_w, cg0_b, cg0_gm, cg0_bt, cg0_mu, cg0_vr, B, H, W);
    conv3x3_f32<16, 8, 1><<<grd, blk, 0, stream>>>(t0, t1, cg1_w, cg1_b, cg1_gm, cg1_bt, cg1_mu, cg1_vr, B, H, W);
    conv3x3_f32<8, 4, 1><<<grd, blk, 0, stream>>>(t1, t2, cg2_w, cg2_b, cg2_gm, cg2_bt, cg2_mu, cg2_vr, B, H, W);
    conv3x3_f32<4, 1, 2><<<grd, blk, 0, stream>>>(t2, g_, cg3_w, cg3_b, cg3_gm, cg3_bt, cg3_mu, cg3_vr, B, H, W);
    // 2) tpn0: 1->64 + ReLU -> f16 NHWC (overwrites t1, which is dead)
    conv1toC_nhwc<64><<<grd, blk, 0, stream>>>(x, x1, tpn0_w, tpn0_b, B, H, W);
    // 3) tpn1: 64->64 WMMA + ReLU -> f16 NHWC (overwrites t0/t2, dead)
    tpn1_wmma_kernel<<<wgrd, 128, 0, stream>>>(x1, B1, tpn1_b, x2);
    // 4) tpn2: 64->1
    tpn2_conv<<<grd, blk, 0, stream>>>(x2, x_, tpn2_w, tpn2_b, B, H, W);
    // 5) c1 on concat(x_, g_)
    c1_conv<<<grd, blk, 0, stream>>>(x_, g_, xc, c1_w, c1_b, B, H, W);
    // 6) df0: 1->32 + ReLU -> f16 NHWC (overwrites x1, dead)
    conv1toC_nhwc<32><<<grd, blk, 0, stream>>>(xc, h32, df0_w, df0_b, B, H, W);
    // 7) df1 WMMA + softmax + 9x9 patch dot -> output
    df1_softmax_dot_kernel<<<wgrd, 128, 0, stream>>>(h32, B2, bias96, xc, out);

    (void)in_sizes; (void)n_in; (void)out_size; (void)ws_size;
}
